// ConvGuidedFilter_14955076124858
// MI455X (gfx1250) — matrix operand order repaired and checked by reading, not compiled, not run on hardware
//
#include <hip/hip_runtime.h>
#include <hip/hip_bf16.h>

typedef _Float16 hv8  __attribute__((ext_vector_type(8)));
typedef _Float16 hv16 __attribute__((ext_vector_type(16)));
typedef float    fv4  __attribute__((ext_vector_type(4)));
typedef float    fv8  __attribute__((ext_vector_type(8)));

#define IMG_H 1024
#define IMG_W 1024
#define NB    4
#define HP    31            // coarse grid 31x31
#define PPB   961           // 31*31
#define MV    3844          // NB*PPB valid coarse pixels
#define MP    3856          // padded to multiple of 16
#define NT_M  241           // MP/16 M-tiles
#define CNT_F 3844.0f
#define BN_EPS 1e-5f

// ---------------------------------------------------------------------------
// Kernel 1: per-32x32-tile sums of (g, s, g*s, g*g) for all 3 channels.
// Grid: NB*32*32 blocks of 256 threads. Each thread loads one float4 (b128)
// per channel per image -> inputs read exactly once at full vector width.
// S layout: [(b*3+c)*4 + stat][32*32] floats, stat: 0=g 1=s 2=gs 3=gg
// ---------------------------------------------------------------------------
__global__ __launch_bounds__(256) void k_tile_sums(const float* __restrict__ guide,
                                                   const float* __restrict__ src,
                                                   float* __restrict__ S) {
  int bid = blockIdx.x;            // 0..4095
  int b   = bid >> 10;
  int t   = bid & 1023;            // tile linear index ty*32+tx
  int ty  = t >> 5, tx = t & 31;
  int tid = threadIdx.x;
  int row0 = ty * 32, col0 = tx * 32;

  // 256 threads x 1 float4 = 1024 pixels (the whole 32x32 tile)
  int pr = tid >> 3;               // row within tile (0..31)
  int pc = (tid & 7) * 4;          // col within tile (0,4,...,28)
  int idx = (row0 + pr) * IMG_W + (col0 + pc);

  float acc[12];
#pragma unroll
  for (int k = 0; k < 12; ++k) acc[k] = 0.0f;

#pragma unroll
  for (int c = 0; c < 3; ++c) {
    const float* gp = guide + (size_t)(b * 3 + c) * IMG_H * IMG_W;
    const float* sp = src   + (size_t)(b * 3 + c) * IMG_H * IMG_W;
    fv4 g4 = *(const fv4*)(gp + idx);
    fv4 s4 = *(const fv4*)(sp + idx);
#pragma unroll
    for (int j = 0; j < 4; ++j) {
      float g = g4[j], s = s4[j];
      acc[c * 4 + 0] += g;
      acc[c * 4 + 1] += s;
      acc[c * 4 + 2] += g * s;
      acc[c * 4 + 3] += g * g;
    }
  }

  // wave32 butterfly reduce, then cross-wave via LDS
#pragma unroll
  for (int k = 0; k < 12; ++k) {
    float v = acc[k];
#pragma unroll
    for (int off = 16; off > 0; off >>= 1) v += __shfl_xor(v, off, 32);
    acc[k] = v;
  }
  __shared__ float red[8][12];
  int lane = tid & 31, wv = tid >> 5;
  if (lane == 0) {
#pragma unroll
    for (int k = 0; k < 12; ++k) red[wv][k] = acc[k];
  }
  __syncthreads();
  if (tid < 12) {
    float s = 0.0f;
#pragma unroll
    for (int w = 0; w < 8; ++w) s += red[w][tid];
    int c = tid >> 2, st = tid & 3;
    S[(((b * 3 + c) * 4 + st) << 10) + t] = s;
  }
}

// ---------------------------------------------------------------------------
// Kernel 2: combine 2x2 tiles -> window stats; emit WMMA A-matrix rows (f16,
// K padded 6->32), and mean_x / mean_y planes. One thread per padded row.
// ---------------------------------------------------------------------------
__global__ __launch_bounds__(256) void k_make_h(const float* __restrict__ S,
                                                _Float16* __restrict__ Hm,
                                                float* __restrict__ mx_,
                                                float* __restrict__ my_) {
  int m = blockIdx.x * blockDim.x + threadIdx.x;
  if (m >= MP) return;
  _Float16* row = Hm + (size_t)m * 32;
  if (m >= MV) {  // zero padding rows (also clears workspace poison)
#pragma unroll
    for (int k = 0; k < 32; ++k) row[k] = (_Float16)0.0f;
    return;
  }
  int b = m / PPB, p = m % PPB;
  int i = p / HP, j = p % HP;
#pragma unroll
  for (int k = 6; k < 32; ++k) row[k] = (_Float16)0.0f;

  const float inv = 1.0f / 4096.0f;  // window count N == 64*64 (VALID padding)
#pragma unroll
  for (int c = 0; c < 3; ++c) {
    const float* base = S + (((b * 3 + c) * 4) << 10) + i * 32 + j;
    float sg  = base[0]      + base[1]      + base[32]     + base[33];
    float ss  = base[1024]   + base[1025]   + base[1056]   + base[1057];
    float sgs = base[2048]   + base[2049]   + base[2080]   + base[2081];
    float sgg = base[3072]   + base[3073]   + base[3104]   + base[3105];
    float mx = sg * inv, my = ss * inv;
    float cov = sgs * inv - mx * my;
    float var = sgg * inv - mx * mx;
    row[c]     = (_Float16)cov;   // h channels 0..2 = cov_xy
    row[3 + c] = (_Float16)var;   // h channels 3..5 = var_x
    mx_[(b * 3 + c) * PPB + p] = mx;
    my_[(b * 3 + c) * PPB + p] = my;
  }
}

// ---------------------------------------------------------------------------
// Kernel 3: entire MLP (conv1x1 -> BN -> relu, x2, conv1x1) in ONE workgroup
// of 8 waves using v_wmma_f32_16x16x32_f16. M = coarse pixels (16/tile),
// N = channels, K = input channels (exactly 32 for conv2/conv3's input).
// BN batch stats via deterministic shuffle+LDS reductions between phases.
// ---------------------------------------------------------------------------
__device__ inline fv8 wmma16(hv16 a, hv16 b, fv8 c) {
  return __builtin_amdgcn_wmma_f32_16x16x32_f16(false, a, false, b, (short)0, c,
                                                false, false);
}

__global__ __launch_bounds__(256) void k_mlp(
    const _Float16* __restrict__ Hm, float* __restrict__ T1, float* __restrict__ T2,
    const float* __restrict__ mx_, const float* __restrict__ my_,
    float* __restrict__ Ac, float* __restrict__ Bc,
    const float* __restrict__ w1, const float* __restrict__ g1, const float* __restrict__ b1,
    const float* __restrict__ w2, const float* __restrict__ g2, const float* __restrict__ b2,
    const float* __restrict__ w3) {
  __shared__ float redS[8][32], redQ[8][32];
  __shared__ float sc1[32], sh1[32], sc2[32], sh2[32];

  int tid  = threadIdx.x;
  int lane = tid & 31;
  int wv   = tid >> 5;      // 8 waves
  int g    = lane >> 4;     // half-wave: selects K-group / M-group
  int lm   = lane & 15;

  // --- build B fragments (K x N, f16) straight from global weights ---------
  // lane layout (16x16x32 f16 B): element e -> B[K = (e<8?0:16) + 8*g + (e&7)][N = lane%16]
  hv16 bf1[2] = {}, bf2[2] = {}, bf3 = {};
#pragma unroll
  for (int e = 0; e < 16; ++e) {
    int k = ((e < 8) ? 0 : 16) + 8 * g + (e & 7);
#pragma unroll
    for (int nt = 0; nt < 2; ++nt) {
      int n = nt * 16 + lm;
      bf1[nt][e] = (_Float16)((k < 6) ? w1[n * 6 + k] : 0.0f);   // w1:[32,6]
      bf2[nt][e] = (_Float16)(w2[n * 32 + k]);                   // w2:[32,32]
    }
    bf3[e] = (_Float16)((lm < 3) ? w3[lm * 32 + k] : 0.0f);      // w3:[3,32]
  }
  __syncthreads();

  // --- phase 1: conv1 (K=32 incl. zero padding), accumulate BN1 stats ------
  float lS[2] = {0.0f, 0.0f}, lQ[2] = {0.0f, 0.0f};
  for (int mt = wv; mt < NT_M; mt += 8) {
    int m0 = mt * 16;
    const _Float16* arow = Hm + (size_t)(m0 + lm) * 32;
    hv8 alo = *(const hv8*)(arow + 8 * g);
    hv8 ahi = *(const hv8*)(arow + 16 + 8 * g);
    hv16 a = __builtin_shufflevector(alo, ahi, 0, 1, 2, 3, 4, 5, 6, 7, 8, 9, 10,
                                     11, 12, 13, 14, 15);
#pragma unroll
    for (int nt = 0; nt < 2; ++nt) {
      fv8 c = {};
      c = wmma16(a, bf1[nt], c);
      int n = nt * 16 + lm;
      float s = 0.0f, q = 0.0f;
#pragma unroll
      for (int r = 0; r < 8; ++r) {
        int gm = m0 + r + 8 * g;                 // D row (padded rows yield 0)
        float v = c[r];
        T1[(size_t)gm * 32 + n] = v;
        s += v; q += v * v;
      }
      lS[nt] += s; lQ[nt] += q;
    }
  }
  // deterministic stats reduce: lane^16 shares same channel n
#pragma unroll
  for (int nt = 0; nt < 2; ++nt) {
    float s = lS[nt] + __shfl_xor(lS[nt], 16, 32);
    float q = lQ[nt] + __shfl_xor(lQ[nt], 16, 32);
    if (lane < 16) { redS[wv][nt * 16 + lane] = s; redQ[wv][nt * 16 + lane] = q; }
  }
  __syncthreads();
  if (tid < 32) {
    float s = 0.0f, q = 0.0f;
#pragma unroll
    for (int w = 0; w < 8; ++w) { s += redS[w][tid]; q += redQ[w][tid]; }
    float mean = s / CNT_F;
    float var  = q / CNT_F - mean * mean;
    float sc   = g1[tid] * rsqrtf(var + BN_EPS);
    sc1[tid] = sc;
    sh1[tid] = b1[tid] - mean * sc;
  }
  __syncthreads();

  // --- phase 2: bn1+relu on the fly -> conv2 (K=32), BN2 stats -------------
  lS[0] = lS[1] = lQ[0] = lQ[1] = 0.0f;
  for (int mt = wv; mt < NT_M; mt += 8) {
    int m0 = mt * 16;
    int mrow = m0 + lm;
    bool valid = mrow < MV;
    const float* trow = T1 + (size_t)mrow * 32;
    hv16 a = {};
#pragma unroll
    for (int e = 0; e < 16; ++e) {
      int k = ((e < 8) ? 0 : 16) + 8 * g + (e & 7);
      float v = valid ? fmaxf(trow[k] * sc1[k] + sh1[k], 0.0f) : 0.0f;
      a[e] = (_Float16)v;
    }
#pragma unroll
    for (int nt = 0; nt < 2; ++nt) {
      fv8 c = {};
      c = wmma16(a, bf2[nt], c);
      int n = nt * 16 + lm;
      float s = 0.0f, q = 0.0f;
#pragma unroll
      for (int r = 0; r < 8; ++r) {
        int gm = m0 + r + 8 * g;
        float v = c[r];
        T2[(size_t)gm * 32 + n] = v;
        s += v; q += v * v;
      }
      lS[nt] += s; lQ[nt] += q;
    }
  }
#pragma unroll
  for (int nt = 0; nt < 2; ++nt) {
    float s = lS[nt] + __shfl_xor(lS[nt], 16, 32);
    float q = lQ[nt] + __shfl_xor(lQ[nt], 16, 32);
    if (lane < 16) { redS[wv][nt * 16 + lane] = s; redQ[wv][nt * 16 + lane] = q; }
  }
  __syncthreads();
  if (tid < 32) {
    float s = 0.0f, q = 0.0f;
#pragma unroll
    for (int w = 0; w < 8; ++w) { s += redS[w][tid]; q += redQ[w][tid]; }
    float mean = s / CNT_F;
    float var  = q / CNT_F - mean * mean;
    float sc   = g2[tid] * rsqrtf(var + BN_EPS);
    sc2[tid] = sc;
    sh2[tid] = b2[tid] - mean * sc;
  }
  __syncthreads();

  // --- phase 3: bn2+relu -> conv3 (N=3 in one N-tile) -> A, b coefficients -
  for (int mt = wv; mt < NT_M; mt += 8) {
    int m0 = mt * 16;
    int mrow = m0 + lm;
    bool valid = mrow < MV;
    const float* trow = T2 + (size_t)mrow * 32;
    hv16 a = {};
#pragma unroll
    for (int e = 0; e < 16; ++e) {
      int k = ((e < 8) ? 0 : 16) + 8 * g + (e & 7);
      float v = valid ? fmaxf(trow[k] * sc2[k] + sh2[k], 0.0f) : 0.0f;
      a[e] = (_Float16)v;
    }
    fv8 c = {};
    c = wmma16(a, bf3, c);
    if (lm < 3) {  // only channels 0..2 are real outputs
#pragma unroll
      for (int r = 0; r < 8; ++r) {
        int gm = m0 + r + 8 * g;
        if (gm < MV) {
          int b = gm / PPB, p = gm % PPB;
          int o = (b * 3 + lm) * PPB + p;
          float Av = c[r];
          Ac[o] = Av;
          Bc[o] = my_[o] - Av * mx_[o];   // b = mean_y - A*mean_x
        }
      }
    }
  }
}

// ---------------------------------------------------------------------------
// Kernel 4: fused bilinear (align_corners=True) upsample of A and b with the
// final A*guide + b. Each thread produces 4 consecutive output pixels (one
// b128 load of guide, one b128 store of out; shared y-interpolation).
// A/b planes are tiny (46 KB) -> L2-resident gathers.
// ---------------------------------------------------------------------------
__global__ __launch_bounds__(256) void k_final(const float* __restrict__ guide,
                                               const float* __restrict__ Ac,
                                               const float* __restrict__ Bc,
                                               float* __restrict__ out) {
  int idx4 = blockIdx.x * blockDim.x + threadIdx.x;   // one float4 per thread
  if (idx4 >= NB * 3 * IMG_H * IMG_W / 4) return;
  int plane = idx4 >> 18;            // (H*W/4) = 2^18 float4 per plane
  int rem   = idx4 & ((1 << 18) - 1);
  int oy  = rem >> 8;                // 256 float4 per row
  int ox0 = (rem & 255) * 4;

  const float scl = 30.0f / 1023.0f;   // (h'-1)/(H-1), align_corners
  float sy = oy * scl;
  int   y0 = (int)sy;
  float wy = sy - (float)y0;
  int   y1 = (y0 + 1 < 31) ? y0 + 1 : 30;

  const float* Ap0 = Ac + plane * PPB + y0 * HP;
  const float* Ap1 = Ac + plane * PPB + y1 * HP;
  const float* Bp0 = Bc + plane * PPB + y0 * HP;
  const float* Bp1 = Bc + plane * PPB + y1 * HP;

  int base = (plane << 20) + oy * IMG_W + ox0;
  fv4 g4 = *(const fv4*)(guide + base);
  fv4 o4;
#pragma unroll
  for (int j = 0; j < 4; ++j) {
    float sx = (float)(ox0 + j) * scl;
    int   x0 = (int)sx;
    float wx = sx - (float)x0;
    int   x1 = (x0 + 1 < 31) ? x0 + 1 : 30;
    float a  = (Ap0[x0] * (1.0f - wx) + Ap0[x1] * wx) * (1.0f - wy) +
               (Ap1[x0] * (1.0f - wx) + Ap1[x1] * wx) * wy;
    float bb = (Bp0[x0] * (1.0f - wx) + Bp0[x1] * wx) * (1.0f - wy) +
               (Bp1[x0] * (1.0f - wx) + Bp1[x1] * wx) * wy;
    o4[j] = a * g4[j] + bb;
  }
  *(fv4*)(out + base) = o4;
}

// ---------------------------------------------------------------------------
extern "C" void kernel_launch(void* const* d_in, const int* in_sizes, int n_in,
                              void* d_out, int out_size, void* d_ws, size_t ws_size,
                              hipStream_t stream) {
  const float* guide = (const float*)d_in[0];
  const float* src   = (const float*)d_in[1];
  // d_in[2] = box_w (all ones) -- unused: VALID padding => N == 64*64 constant
  const float* w1 = (const float*)d_in[3];
  const float* g1 = (const float*)d_in[4];
  const float* b1 = (const float*)d_in[5];
  const float* w2 = (const float*)d_in[6];
  const float* g2 = (const float*)d_in[7];
  const float* b2 = (const float*)d_in[8];
  const float* w3 = (const float*)d_in[9];
  float* out = (float*)d_out;

  char* wsb = (char*)d_ws;
  size_t off = 0;
  auto take = [&](size_t bytes) -> void* {
    void* p = wsb + off;
    off += (bytes + 255) & ~(size_t)255;
    return p;
  };
  float*    S  = (float*)take((size_t)NB * 3 * 4 * 1024 * 4);   // tile sums
  _Float16* Hm = (_Float16*)take((size_t)MP * 32 * 2);          // conv1 A-matrix
  float*    T1 = (float*)take((size_t)MP * 32 * 4);             // conv1 raw out
  float*    T2 = (float*)take((size_t)MP * 32 * 4);             // conv2 raw out
  float*    mx = (float*)take((size_t)MV * 4);
  float*    my = (float*)take((size_t)MV * 4);
  float*    Ac = (float*)take((size_t)MV * 4);
  float*    Bc = (float*)take((size_t)MV * 4);
  (void)ws_size; (void)in_sizes; (void)n_in; (void)out_size;

  k_tile_sums<<<NB * 32 * 32, 256, 0, stream>>>(guide, src, S);
  k_make_h<<<(MP + 255) / 256, 256, 0, stream>>>(S, Hm, mx, my);
  k_mlp<<<1, 256, 0, stream>>>(Hm, T1, T2, mx, my, Ac, Bc,
                               w1, g1, b1, w2, g2, b2, w3);
  k_final<<<(NB * 3 * IMG_H * IMG_W / 4 + 255) / 256, 256, 0, stream>>>(guide, Ac, Bc, out);
}
